// EdgeDecoder_10359461118099
// MI455X (gfx1250) — matrix operand order repaired and checked by reading, not compiled, not run on hardware
//
#include <hip/hip_runtime.h>
#include <hip/hip_bf16.h>

// ---------------------------------------------------------------------------
// EdgeDecoder: out[e] = W2 . relu(W1 . [h[src[e]] ; h[dst[e]]] + b1) + b2
// v3: 32 edges per block iteration, 4 waves x 32 output rows each (two
// A-tiles per wave) so every 1 KB LDS B-fragment read feeds TWO WMMAs.
// B fragments are software-pipelined with TWO live buffers (no copy between
// them) so ds_load_b128 of chunk c+2 overlaps the WMMAs of chunk c+1 and the
// compiler cannot coalesce the buffers into one register set.
// ---------------------------------------------------------------------------

typedef __attribute__((ext_vector_type(16))) __bf16 v16bf;
typedef __attribute__((ext_vector_type(8)))  __bf16 v8bf;
typedef __attribute__((ext_vector_type(8)))  float  v8f;
typedef __attribute__((ext_vector_type(4)))  float  v4f;

#define HD        128
#define XCOLS     256            // 2*HD features per edge row
#define TILE_E    32             // edges per block iteration (2 x WMMA N)
#define XSTRIDE   (XCOLS + 8)    // +16B pad: LDS bank spread for b128 reads

__device__ __forceinline__ __bf16 f2bf(float f) { return (__bf16)f; }

__device__ __forceinline__ v16bf ldB(const __bf16* base, int c) {
    const __bf16* bp = base + c * 32;
    const v8bf lo  = *(const v8bf*)(bp + 0);
    const v8bf hi8 = *(const v8bf*)(bp + 16);
    return __builtin_shufflevector(lo, hi8,
        0, 1, 2, 3, 4, 5, 6, 7, 8, 9, 10, 11, 12, 13, 14, 15);
}

#define WMMA_BF16(A, B, C) \
    __builtin_amdgcn_wmma_f32_16x16x32_bf16(false, (A), false, (B), (short)0, (C), false, false)

__global__ __launch_bounds__(256, 1)
void edge_decoder_wmma(const int* __restrict__ src,
                       const int* __restrict__ dst,
                       const float* __restrict__ h,
                       const float* __restrict__ W1,   // [128, 256] row-major
                       const float* __restrict__ b1,   // [128]
                       const float* __restrict__ W2,   // [128] (row of [1,128])
                       const float* __restrict__ b2,   // [1]
                       float* __restrict__ out,        // [E]
                       int E, int ntiles)
{
    __shared__ __align__(16) __bf16 Xs[TILE_E * XSTRIDE];  // ~16.9 KB
    __shared__ float parts[8 * 32];

    const int tid   = threadIdx.x;
    const int lane  = tid & 31;
    const int wave  = tid >> 5;
    const int wl    = wave & 3;    // which 32-row W1 slice this wave owns
    const int grp   = wave >> 2;   // which 16-edge sub-tile this wave computes
    const int row16 = lane & 15;
    const int hi    = lane >> 4;

    // ---- Preload TWO W1 A-tiles per wave as bf16 fragments (VGPR-resident).
    // A 16x32 bf16 layout: lane l<16 -> row M=l, elems K{0..7,16..23};
    //                      lane l>=16 -> row M=l-16, elems K{8..15,24..31}.
    const int arow0 = wl * 32 + row16;
    const int arow1 = arow0 + 16;
    v16bf aW0[8], aW1[8];
#pragma unroll
    for (int c = 0; c < 8; ++c) {
        const float* p0 = W1 + arow0 * XCOLS + c * 32 + hi * 8;
        const float* p1 = W1 + arow1 * XCOLS + c * 32 + hi * 8;
        v16bf a0, a1;
#pragma unroll
        for (int i = 0; i < 8; ++i) {
            a0[i]     = f2bf(p0[i]);
            a0[8 + i] = f2bf(p0[16 + i]);
            a1[i]     = f2bf(p1[i]);
            a1[8 + i] = f2bf(p1[16 + i]);
        }
        aW0[c] = a0;
        aW1[c] = a1;
    }

    // ---- Per-lane bias / W2 slices for both M-tiles.
    // D layout: VGPR r, lanes<16 -> M=r, lanes>=16 -> M=8+r.
    const int mb0 = wl * 32 + hi * 8;
    v8f bias0, bias1, w20, w21;
#pragma unroll
    for (int r = 0; r < 8; ++r) {
        bias0[r] = b1[mb0 + r];
        w20[r]   = W2[mb0 + r];
        bias1[r] = b1[mb0 + 16 + r];
        w21[r]   = W2[mb0 + 16 + r];
    }
    const float b2v = b2[0];

    // Staging role: each thread loads 32 consecutive floats of one X row.
    const int srow = tid >> 3;          // 0..31 edge row
    const int sseg = (tid & 7) * 32;    // col base 0..224 (never crosses 128)

    for (int tile = blockIdx.x; tile < ntiles; tile += gridDim.x) {
        // ---------------- Stage 32-edge X tile (fp32 gather -> bf16 LDS) --
        {
            int e  = tile * TILE_E + srow;
            int eL = e < E ? e : (E - 1);
            const int idx  = (sseg < HD) ? src[eL] : dst[eL];
            const int hoff = sseg & (HD - 1);
            const float* hp = h + (long)idx * HD + hoff;
            __bf16* xp = &Xs[srow * XSTRIDE + sseg];
#pragma unroll
            for (int q = 0; q < 4; ++q) {
                const v4f fa = *(const v4f*)(hp + q * 8 + 0);
                const v4f fb = *(const v4f*)(hp + q * 8 + 4);
                v8bf v;
#pragma unroll
                for (int i = 0; i < 4; ++i) { v[i] = f2bf(fa[i]); v[4 + i] = f2bf(fb[i]); }
                *(v8bf*)(xp + q * 8) = v;
            }
        }
        __syncthreads();

        // ---------------- 16x WMMA, 2-deep pipelined B fragments ----------
        // B 32x16 bf16 layout: lane n<16 -> col n, K=k0+{0..7,16..23};
        //                      lane n>=16 -> col n-16, K=k0+{8..15,24..31}.
        const __bf16* xrow = &Xs[(grp * 16 + row16) * XSTRIDE + hi * 8];
        v8f acc0 = bias0, acc1 = bias1;
        v16bf bA = ldB(xrow, 0);
        v16bf bB = ldB(xrow, 1);
#pragma unroll
        for (int c = 0; c < 8; c += 2) {
            acc0 = WMMA_BF16(aW0[c], bA, acc0);
            acc1 = WMMA_BF16(aW1[c], bA, acc1);
            if (c + 2 < 8) bA = ldB(xrow, c + 2);   // overlaps bB's WMMAs
            acc0 = WMMA_BF16(aW0[c + 1], bB, acc0);
            acc1 = WMMA_BF16(aW1[c + 1], bB, acc1);
            if (c + 3 < 8) bB = ldB(xrow, c + 3);   // overlaps bA's WMMAs
        }

        // ---------------- relu + W2 dot -> per-lane partial ---------------
        float p = 0.f;
#pragma unroll
        for (int r = 0; r < 8; ++r) {
            p = fmaf(fmaxf(acc0[r], 0.f), w20[r], p);
            p = fmaf(fmaxf(acc1[r], 0.f), w21[r], p);
        }
        parts[wave * 32 + lane] = p;
        __syncthreads();

        // ---------------- reduce 8 partials per edge, store ---------------
        if (tid < 32) {
            const int g = tid >> 4;      // edge sub-tile
            const int n = tid & 15;      // edge within sub-tile
            float s = b2v;
#pragma unroll
            for (int w = 0; w < 4; ++w) {
                const int ww = g * 4 + w;
                s += parts[ww * 32 + n] + parts[ww * 32 + n + 16];
            }
            const int e = tile * TILE_E + g * 16 + n;
            if (e < E) out[e] = s;
        }
        // Next iteration's post-staging __syncthreads orders parts[] reuse.
    }
}

extern "C" void kernel_launch(void* const* d_in, const int* in_sizes, int n_in,
                              void* d_out, int out_size, void* d_ws, size_t ws_size,
                              hipStream_t stream) {
    const int*   edges = (const int*)d_in[0];   // [2, E]
    const float* h     = (const float*)d_in[1]; // [N, 128]
    const float* W1    = (const float*)d_in[2]; // [128, 256]
    const float* b1    = (const float*)d_in[3]; // [128]
    const float* W2    = (const float*)d_in[4]; // [1, 128]
    const float* b2    = (const float*)d_in[5]; // [1]
    float*       out   = (float*)d_out;

    const int E      = in_sizes[0] / 2;
    const int ntiles = (E + TILE_E - 1) / TILE_E;
    int blocks = ntiles < 2048 ? ntiles : 2048;

    edge_decoder_wmma<<<blocks, 256, 0, stream>>>(
        edges, edges + E, h, W1, b1, W2, b2, out, E, ntiles);
}
